// RULE_Encoder_5806795784688
// MI455X (gfx1250) — compile-verified, matrix operand order checked by reading, not run on hardware
//
#include <hip/hip_runtime.h>

#define NODE   30000
#define RELN   1000
#define EDGES  300000
#define DIM    128
#define OUTD   256
#define PROXYN 64
#define WPB    8   // waves per block (256 threads, wave32)

typedef __attribute__((ext_vector_type(16))) _Float16 v16h;
typedef __attribute__((ext_vector_type(8)))  float    v8f;

static inline int cdiv(long long a, long long b) { return (int)((a + b - 1) / b); }

// ---------------- wave32 helpers ----------------
__device__ __forceinline__ float waveSum(float x) {
#pragma unroll
  for (int off = 16; off > 0; off >>= 1) x += __shfl_xor(x, off, 32);
  return x;
}
__device__ __forceinline__ float waveMax(float x) {
#pragma unroll
  for (int off = 16; off > 0; off >>= 1) x = fmaxf(x, __shfl_xor(x, off, 32));
  return x;
}
// float atomic max via signed/unsigned integer ordering trick
__device__ __forceinline__ void atomicMaxF(float* addr, float v) {
  if (v >= 0.0f) atomicMax((int*)addr, __float_as_int(v));
  else           atomicMin((unsigned int*)addr, (unsigned int)__float_as_int(v));
}

// ---------------- elementwise utility kernels ----------------
__global__ void k_fill(float* __restrict__ p, float v, int n) {
  int i = blockIdx.x * blockDim.x + threadIdx.x;
  if (i < n) p[i] = v;
}
__global__ void k_tanh_ip(float* __restrict__ p, int n) {
  int i = blockIdx.x * blockDim.x + threadIdx.x;
  if (i < n) p[i] = tanhf(p[i]);
}
__global__ void k_sub_from(const float* __restrict__ a, float* __restrict__ b, int n) {
  int i = blockIdx.x * blockDim.x + threadIdx.x;
  if (i < n) b[i] = a[i] - b[i];   // b = a - b  (pf = out - p_att@proxy)
}

// ---------------- sparse mean (avg aggregator) ----------------
__global__ void k_deg(const int* __restrict__ rows, float* __restrict__ deg, int n) {
  int i = blockIdx.x * blockDim.x + threadIdx.x;
  if (i < n) atomicAdd(&deg[rows[i]], 1.0f);
}
// one 32-lane group per edge; each lane handles 4 contiguous dims (float4)
__global__ void k_scatter_mean(const int* __restrict__ rows, const int* __restrict__ cols,
                               const float* __restrict__ emb, const float* __restrict__ deg,
                               float* __restrict__ acc, int nEdges) {
  int t = blockIdx.x * blockDim.x + threadIdx.x;
  int e = t >> 5;
  if (e >= nEdges) return;
  int c = t & 31;
  int r = rows[e];
  float dg = deg[r];
  float w = dg > 0.0f ? 1.0f / dg : 0.0f;
  float4 v = ((const float4*)(emb + (size_t)cols[e] * DIM))[c];
  float* dst = acc + (size_t)r * DIM + c * 4;
  atomicAdd(dst + 0, v.x * w);
  atomicAdd(dst + 1, v.y * w);
  atomicAdd(dst + 2, v.z * w);
  atomicAdd(dst + 3, v.w * w);
}

// ---------------- WMMA fragment loaders ----------------
// 16-bit fragment layout: lane<16 holds K {0..7, 16..23}, lane>=16 holds {8..15, 24..31}
// relative to chunk base k0. Both 8-groups are contiguous -> four b128 loads.
__device__ __forceinline__ v16h frag_from_row(const float* __restrict__ rowp, int k0, int hi) {
  const float4* p0 = (const float4*)(rowp + k0 + 8 * hi);
  const float4* p1 = (const float4*)(rowp + k0 + 16 + 8 * hi);
  float4 x0 = p0[0], x1 = p0[1], x2 = p1[0], x3 = p1[1];
  v16h a;
  a[0]  = (_Float16)x0.x; a[1]  = (_Float16)x0.y; a[2]  = (_Float16)x0.z; a[3]  = (_Float16)x0.w;
  a[4]  = (_Float16)x1.x; a[5]  = (_Float16)x1.y; a[6]  = (_Float16)x1.z; a[7]  = (_Float16)x1.w;
  a[8]  = (_Float16)x2.x; a[9]  = (_Float16)x2.y; a[10] = (_Float16)x2.z; a[11] = (_Float16)x2.w;
  a[12] = (_Float16)x3.x; a[13] = (_Float16)x3.y; a[14] = (_Float16)x3.z; a[15] = (_Float16)x3.w;
  return a;
}
// masked tail: clamp index (stay in-bounds) + select 0 -> v_cndmask, no exec branching
__device__ __forceinline__ v16h frag_from_row_tail(const float* __restrict__ rowp,
                                                   int k0, int hi, int K) {
  v16h a;
#pragma unroll
  for (int h = 0; h < 16; ++h) {
    int kk = k0 + ((h < 8) ? h : (h + 8)) + hi * 8;
    int kc = kk < K ? kk : K - 1;
    float v = rowp[kc];
    a[h] = (kk < K) ? (_Float16)v : (_Float16)0.0f;
  }
  return a;
}
// B fragment from [K,N] row-major (column n), strided along K
__device__ __forceinline__ v16h frag_from_colB(const float* __restrict__ B, int ldb,
                                               int n, int k0, int hi) {
  v16h b;
#pragma unroll
  for (int h = 0; h < 16; ++h) {
    int kk = k0 + ((h < 8) ? h : (h + 8)) + hi * 8;
    b[h] = (_Float16)B[(size_t)kk * ldb + n];
  }
  return b;
}
__device__ __forceinline__ v16h frag_from_colB_tail(const float* __restrict__ B, int ldb,
                                                    int n, int k0, int hi, int K) {
  v16h b;
#pragma unroll
  for (int h = 0; h < 16; ++h) {
    int kk = k0 + ((h < 8) ? h : (h + 8)) + hi * 8;
    int kc = kk < K ? kk : K - 1;
    float v = B[(size_t)kc * ldb + n];
    b[h] = (kk < K) ? (_Float16)v : (_Float16)0.0f;
  }
  return b;
}

// ---------------- WMMA GEMM: C[M,N] = A[M,K] * B (+bias) ----------------
// BT=true : B is W[N,K] row-major (computes A @ W^T)
// BT=false: B is G[K,N] row-major (computes A @ G)
// One wave computes a 16x64 strip. Per k-chunk: load A + four independent B
// fragments (b0..b3), then issue 4 back-to-back WMMAs. Independent B registers
// avoid the WMMA->VALU WAR hazard NOPs between consecutive matrix ops.
template <bool BT>
__global__ void __launch_bounds__(256) k_wmma_gemm(
    const float* __restrict__ A, int lda,
    const float* __restrict__ B, int ldb,
    const float* __restrict__ bias,
    float* __restrict__ C, int ldc,
    int M, int N, int K) {
  const int lane   = threadIdx.x & 31;
  const int wave   = threadIdx.x >> 5;
  const int tilesN = N >> 6;
  const int tile   = blockIdx.x * WPB + wave;
  const int tm     = tile / tilesN;
  const int tn     = tile % tilesN;
  if (tm * 16 >= M) return;          // wave-uniform: EXEC stays all-ones for WMMA
  const int hi = lane >> 4;          // half-wave select
  const int l  = lane & 15;

  v8f acc0 = {}, acc1 = {}, acc2 = {}, acc3 = {};
  const float* Arow = A + (size_t)(tm * 16 + l) * lda;
  const float* Brow = BT ? (B + (size_t)(tn * 64 + l) * ldb) : nullptr;
  const int nBase = tn * 64 + l;

  const int kMain = K & ~31;
  for (int k0 = 0; k0 < kMain; k0 += 32) {
    // WGP-scope prefetch (locality 3): pull next A chunk into all cache levels
    if (k0 + 64 <= kMain) __builtin_prefetch(Arow + k0 + 64, 0, 3);
    v16h a = frag_from_row(Arow, k0, hi);
    v16h b0, b1, b2, b3;
    if (BT) {
      b0 = frag_from_row(Brow, k0, hi);
      b1 = frag_from_row(Brow + (size_t)16 * ldb, k0, hi);
      b2 = frag_from_row(Brow + (size_t)32 * ldb, k0, hi);
      b3 = frag_from_row(Brow + (size_t)48 * ldb, k0, hi);
    } else {
      b0 = frag_from_colB(B, ldb, nBase +  0, k0, hi);
      b1 = frag_from_colB(B, ldb, nBase + 16, k0, hi);
      b2 = frag_from_colB(B, ldb, nBase + 32, k0, hi);
      b3 = frag_from_colB(B, ldb, nBase + 48, k0, hi);
    }
    acc0 = __builtin_amdgcn_wmma_f32_16x16x32_f16(false, a, false, b0, (short)0, acc0, false, false);
    acc1 = __builtin_amdgcn_wmma_f32_16x16x32_f16(false, a, false, b1, (short)0, acc1, false, false);
    acc2 = __builtin_amdgcn_wmma_f32_16x16x32_f16(false, a, false, b2, (short)0, acc2, false, false);
    acc3 = __builtin_amdgcn_wmma_f32_16x16x32_f16(false, a, false, b3, (short)0, acc3, false, false);
  }
  if (kMain < K) {  // single masked tail chunk
    v16h a = frag_from_row_tail(Arow, kMain, hi, K);
    v16h b0, b1, b2, b3;
    if (BT) {
      b0 = frag_from_row_tail(Brow, kMain, hi, K);
      b1 = frag_from_row_tail(Brow + (size_t)16 * ldb, kMain, hi, K);
      b2 = frag_from_row_tail(Brow + (size_t)32 * ldb, kMain, hi, K);
      b3 = frag_from_row_tail(Brow + (size_t)48 * ldb, kMain, hi, K);
    } else {
      b0 = frag_from_colB_tail(B, ldb, nBase +  0, kMain, hi, K);
      b1 = frag_from_colB_tail(B, ldb, nBase + 16, kMain, hi, K);
      b2 = frag_from_colB_tail(B, ldb, nBase + 32, kMain, hi, K);
      b3 = frag_from_colB_tail(B, ldb, nBase + 48, kMain, hi, K);
    }
    acc0 = __builtin_amdgcn_wmma_f32_16x16x32_f16(false, a, false, b0, (short)0, acc0, false, false);
    acc1 = __builtin_amdgcn_wmma_f32_16x16x32_f16(false, a, false, b1, (short)0, acc1, false, false);
    acc2 = __builtin_amdgcn_wmma_f32_16x16x32_f16(false, a, false, b2, (short)0, acc2, false, false);
    acc3 = __builtin_amdgcn_wmma_f32_16x16x32_f16(false, a, false, b3, (short)0, acc3, false, false);
  }
  // C layout: lane group hi, VGPR r -> row M = r + 8*hi, col N = lane&15
  v8f accs[4] = {acc0, acc1, acc2, acc3};
#pragma unroll
  for (int j = 0; j < 4; ++j) {
    int n = tn * 64 + j * 16 + l;
    float bv = bias ? bias[n] : 0.0f;
#pragma unroll
    for (int r = 0; r < 8; ++r) {
      int m = tm * 16 + r + 8 * hi;
      C[(size_t)m * ldc + n] = accs[j][r] + bv;
    }
  }
}

template <bool BT>
static void launch_gemm(const float* A, int lda, const float* B, int ldb,
                        const float* bias, float* C, int ldc,
                        int M, int N, int K, hipStream_t s) {
  int tiles = cdiv(M, 16) * (N >> 6);
  k_wmma_gemm<BT><<<cdiv(tiles, WPB), 256, 0, s>>>(A, lda, B, ldb, bias, C, ldc, M, N, K);
}

// ---------------- nr_gat edge kernels ----------------
// ri0 == arange(E) in the reference, so tri_rel[e] = l2norm(rel_emb[ri1[e]] * rv[e])
// is recomputed on the fly (saves materializing a 154 MB [E,128] tensor).

// pass A: att logit per edge + segment max
__global__ void k_edge_att(const int* __restrict__ row, const int* __restrict__ ri1,
                           const float* __restrict__ rv, const float* __restrict__ relw,
                           const float* __restrict__ attn,
                           float* __restrict__ attE, float* __restrict__ segm, int nEdges) {
  int lane = threadIdx.x & 31;
  int e = blockIdx.x * WPB + (threadIdx.x >> 5);
  if (e >= nEdges) return;
  float s = rv[e];
  float4 t = ((const float4*)(relw + (size_t)ri1[e] * DIM))[lane];
  t.x *= s; t.y *= s; t.z *= s; t.w *= s;
  float ss = waveSum(t.x * t.x + t.y * t.y + t.z * t.z + t.w * t.w);
  float inv = 1.0f / fmaxf(sqrtf(ss), 1e-12f);
  float4 a = ((const float4*)attn)[lane];
  float d = waveSum((t.x * a.x + t.y * a.y + t.z * a.z + t.w * a.w) * inv);
  if (lane == 0) {
    attE[e] = d;
    atomicMaxF(&segm[row[e]], d);
  }
}
// pass B: exp(x - max) and segment sum
__global__ void k_edge_expsum(const int* __restrict__ row, float* __restrict__ attE,
                              const float* __restrict__ segm, float* __restrict__ segs,
                              int nEdges) {
  int e = blockIdx.x * blockDim.x + threadIdx.x;
  if (e >= nEdges) return;
  float ex = expf(attE[e] - segm[row[e]]);
  attE[e] = ex;
  atomicAdd(&segs[row[e]], ex);
}
// pass C: Householder-reflect neighbor features by tri_rel, weight by softmax, scatter-add
__global__ void k_edge_scatter(const int* __restrict__ row, const int* __restrict__ col,
                               const int* __restrict__ ri1, const float* __restrict__ rv,
                               const float* __restrict__ relw, const float* __restrict__ feats,
                               const float* __restrict__ attE, const float* __restrict__ segs,
                               float* __restrict__ newf, int nEdges) {
  int lane = threadIdx.x & 31;
  int e = blockIdx.x * WPB + (threadIdx.x >> 5);
  if (e >= nEdges) return;
  float s = rv[e];
  float4 t = ((const float4*)(relw + (size_t)ri1[e] * DIM))[lane];
  t.x *= s; t.y *= s; t.z *= s; t.w *= s;
  float ss = waveSum(t.x * t.x + t.y * t.y + t.z * t.z + t.w * t.w);
  float inv = 1.0f / fmaxf(sqrtf(ss), 1e-12f);
  t.x *= inv; t.y *= inv; t.z *= inv; t.w *= inv;
  float4 n = ((const float4*)(feats + (size_t)col[e] * DIM))[lane];
  float d = waveSum(n.x * t.x + n.y * t.y + n.z * t.z + n.w * t.w);
  float c2 = 2.0f * d;
  n.x -= c2 * t.x; n.y -= c2 * t.y; n.z -= c2 * t.z; n.w -= c2 * t.w;
  int r = row[e];
  float w = attE[e] / segs[r];
  float* dst = newf + (size_t)r * DIM + lane * 4;
  atomicAdd(dst + 0, n.x * w);
  atomicAdd(dst + 1, n.y * w);
  atomicAdd(dst + 2, n.z * w);
  atomicAdd(dst + 3, n.w * w);
}
// out = concat(feats, tanh(new_feats)) [+ img]
__global__ void k_build_out(const float* __restrict__ feats, const float* __restrict__ newf,
                            const float* __restrict__ img, float* __restrict__ outb, int n) {
  int i = blockIdx.x * blockDim.x + threadIdx.x;
  if (i >= n) return;
  int node = i >> 8, d = i & 255;
  float v = (d < DIM) ? feats[(size_t)node * DIM + d]
                      : tanhf(newf[(size_t)node * DIM + (d - DIM)]);
  if (img) v += img[i];
  outb[i] = v;
}
// row-wise l2 normalize (wave per row)
__global__ void k_l2norm_rows(const float* __restrict__ src, float* __restrict__ dst,
                              int rows, int cols) {
  int lane = threadIdx.x & 31;
  int r = blockIdx.x * WPB + (threadIdx.x >> 5);
  if (r >= rows) return;
  const float* s = src + (size_t)r * cols;
  float ss = 0.0f;
  for (int d = lane; d < cols; d += 32) { float v = s[d]; ss += v * v; }
  ss = waveSum(ss);
  float inv = 1.0f / fmaxf(sqrtf(ss), 1e-12f);
  float* o = dst + (size_t)r * cols;
  for (int d = lane; d < cols; d += 32) o[d] = s[d] * inv;
}
// row-wise softmax (wave per row)
__global__ void k_softmax_rows(float* __restrict__ p, int rows, int cols) {
  int lane = threadIdx.x & 31;
  int r = blockIdx.x * WPB + (threadIdx.x >> 5);
  if (r >= rows) return;
  float* q = p + (size_t)r * cols;
  float m = -3.402823466e38f;
  for (int d = lane; d < cols; d += 32) m = fmaxf(m, q[d]);
  m = waveMax(m);
  float s = 0.0f;
  for (int d = lane; d < cols; d += 32) { float ex = expf(q[d] - m); q[d] = ex; s += ex; }
  s = waveSum(s);
  float inv = 1.0f / s;
  for (int d = lane; d < cols; d += 32) q[d] *= inv;
}
// gate = sigmoid(pfg + bias); dst = gate*out + (1-gate)*pf
__global__ void k_gate(const float* __restrict__ outb, const float* __restrict__ pf,
                       const float* __restrict__ pfg, const float* __restrict__ bias,
                       float* __restrict__ dst, int n) {
  int i = blockIdx.x * blockDim.x + threadIdx.x;
  if (i >= n) return;
  int c = i & 255;
  float g = 1.0f / (1.0f + expf(-(pfg[i] + bias[c])));
  dst[i] = g * outb[i] + (1.0f - g) * pf[i];
}

// ---------------- host-side nr_gat pipeline ----------------
static void run_nr_gat(const float* feats, const float* relw,
                       const int* row, const int* col, const int* ri1, const float* rv,
                       const float* attn, const float* proxy, const float* gatew,
                       const float* biasv, const float* img, float* dst,
                       float* segm, float* segs, float* attE, float* newf,
                       float* outb, float* lnout, float* lnproxy,
                       float* patt, float* pf, float* pfg, hipStream_t s) {
  const int NO = NODE * OUTD;
  k_fill<<<cdiv(NODE, 256), 256, 0, s>>>(segm, -3.402823466e38f, NODE);
  k_fill<<<cdiv(NODE, 256), 256, 0, s>>>(segs, 0.0f, NODE);
  k_fill<<<cdiv(NODE * DIM, 256), 256, 0, s>>>(newf, 0.0f, NODE * DIM);
  k_edge_att<<<cdiv(EDGES, WPB), 256, 0, s>>>(row, ri1, rv, relw, attn, attE, segm, EDGES);
  k_edge_expsum<<<cdiv(EDGES, 256), 256, 0, s>>>(row, attE, segm, segs, EDGES);
  k_edge_scatter<<<cdiv(EDGES, WPB), 256, 0, s>>>(row, col, ri1, rv, relw, feats, attE, segs,
                                                  newf, EDGES);
  k_build_out<<<cdiv(NO, 256), 256, 0, s>>>(feats, newf, img, outb, NO);
  k_l2norm_rows<<<cdiv(NODE, WPB), 256, 0, s>>>(outb, lnout, NODE, OUTD);
  k_l2norm_rows<<<cdiv(PROXYN, WPB), 256, 0, s>>>(proxy, lnproxy, PROXYN, OUTD);
  // p_att = lnout @ lnproxy^T  (WMMA, BT)
  launch_gemm<true>(lnout, OUTD, lnproxy, OUTD, nullptr, patt, PROXYN,
                    NODE, PROXYN, OUTD, s);
  k_softmax_rows<<<cdiv(NODE, WPB), 256, 0, s>>>(patt, NODE, PROXYN);
  // pf = p_att @ proxy  (WMMA), then pf = out - pf
  launch_gemm<false>(patt, PROXYN, proxy, OUTD, nullptr, pf, OUTD,
                     NODE, OUTD, PROXYN, s);
  k_sub_from<<<cdiv(NO, 256), 256, 0, s>>>(outb, pf, NO);
  // pfg = pf @ gate_k (WMMA; bias folded into sigmoid kernel)
  launch_gemm<false>(pf, OUTD, gatew, OUTD, nullptr, pfg, OUTD,
                     NODE, OUTD, OUTD, s);
  k_gate<<<cdiv(NO, 256), 256, 0, s>>>(outb, pf, pfg, biasv, dst, NO);
}

// ---------------- entry point ----------------
extern "C" void kernel_launch(void* const* d_in, const int* in_sizes, int n_in,
                              void* d_out, int out_size, void* d_ws, size_t ws_size,
                              hipStream_t stream) {
  (void)in_sizes; (void)n_in; (void)out_size; (void)ws_size;
  // inputs (setup_inputs order); d_in[0] = mask (unused by reference math)
  const float* imgf     = (const float*)d_in[1];
  const float* attf     = (const float*)d_in[2];
  const float* namef    = (const float*)d_in[3];
  const float* charf    = (const float*)d_in[4];
  const int*   adj      = (const int*)d_in[5];   // [2,E]: rows, cols
  const int*   r_index  = (const int*)d_in[6];   // ri0=arange(E) (unused), ri1
  const float* rval     = (const float*)d_in[7];
  const int*   rel_adj  = (const int*)d_in[8];
  const int*   ent_adj  = (const int*)d_in[9];
  const float* ent_emb  = (const float*)d_in[10];
  const float* relw     = (const float*)d_in[11];
  const float* img_fc_w = (const float*)d_in[12];
  const float* img_fc_b = (const float*)d_in[13];
  const float* att_fc_w = (const float*)d_in[14];
  const float* att_fc_b = (const float*)d_in[15];
  const float* nam_fc_w = (const float*)d_in[16];
  const float* nam_fc_b = (const float*)d_in[17];
  const float* chr_fc_w = (const float*)d_in[18];
  const float* chr_fc_b = (const float*)d_in[19];
  const float* e_attn   = (const float*)d_in[20];
  const float* e_proxy  = (const float*)d_in[21];
  const float* e_gate   = (const float*)d_in[22];
  const float* e_bias   = (const float*)d_in[23];
  const float* r_attn   = (const float*)d_in[24];
  const float* r_proxy  = (const float*)d_in[25];
  const float* r_gate   = (const float*)d_in[26];
  const float* r_bias   = (const float*)d_in[27];

  const int* adj_row = adj;
  const int* adj_col = adj + EDGES;
  const int* ri1     = r_index + EDGES;

  float* out = (float*)d_out;
  float* out_gph  = out + 0 * (size_t)NODE * OUTD;
  float* out_img  = out + 1 * (size_t)NODE * OUTD;
  float* out_rel  = out + 2 * (size_t)NODE * OUTD;
  float* out_att  = out + 3 * (size_t)NODE * OUTD;
  float* out_name = out + 4 * (size_t)NODE * OUTD;
  float* out_char = out + 5 * (size_t)NODE * OUTD;

  // workspace carve-up (~178 MB of f32)
  float* ws = (float*)d_ws;
  size_t off = 0;
  auto alloc = [&](size_t n) { float* p = ws + off; off += n; return p; };
  float* ent_feat = alloc((size_t)NODE * DIM);
  float* rel_feat = alloc((size_t)NODE * DIM);
  float* newf     = alloc((size_t)NODE * DIM);
  float* deg      = alloc(NODE);
  float* segm     = alloc(NODE);
  float* segs     = alloc(NODE);
  float* attE     = alloc(EDGES);
  float* outb     = alloc((size_t)NODE * OUTD);
  float* lnout    = alloc((size_t)NODE * OUTD);
  float* pf       = alloc((size_t)NODE * OUTD);
  float* pfg      = alloc((size_t)NODE * OUTD);
  float* patt     = alloc((size_t)NODE * PROXYN);
  float* lnproxy  = alloc((size_t)PROXYN * OUTD);

  // ---- sparse_mean(ent_adj, ent_emb) -> tanh -> ent_feat ----
  k_fill<<<cdiv(NODE, 256), 256, 0, stream>>>(deg, 0.0f, NODE);
  k_deg<<<cdiv(EDGES, 256), 256, 0, stream>>>(ent_adj, deg, EDGES);
  k_fill<<<cdiv(NODE * DIM, 256), 256, 0, stream>>>(ent_feat, 0.0f, NODE * DIM);
  k_scatter_mean<<<cdiv(EDGES * 32, 256), 256, 0, stream>>>(ent_adj, ent_adj + EDGES,
                                                            ent_emb, deg, ent_feat, EDGES);
  k_tanh_ip<<<cdiv(NODE * DIM, 256), 256, 0, stream>>>(ent_feat, NODE * DIM);

  // ---- sparse_mean(rel_adj, rel_emb_w) -> tanh -> rel_feat ----
  k_fill<<<cdiv(NODE, 256), 256, 0, stream>>>(deg, 0.0f, NODE);
  k_deg<<<cdiv(EDGES, 256), 256, 0, stream>>>(rel_adj, deg, EDGES);
  k_fill<<<cdiv(NODE * DIM, 256), 256, 0, stream>>>(rel_feat, 0.0f, NODE * DIM);
  k_scatter_mean<<<cdiv(EDGES * 32, 256), 256, 0, stream>>>(rel_adj, rel_adj + EDGES,
                                                            relw, deg, rel_feat, EDGES);
  k_tanh_ip<<<cdiv(NODE * DIM, 256), 256, 0, stream>>>(rel_feat, NODE * DIM);

  // ---- dense FC heads (WMMA, A @ W^T + b) ----
  launch_gemm<true>(imgf, 2048, img_fc_w, 2048, img_fc_b, out_img, OUTD, NODE, OUTD, 2048, stream);
  launch_gemm<true>(attf, 1000, att_fc_w, 1000, att_fc_b, out_att, OUTD, NODE, OUTD, 1000, stream);
  launch_gemm<true>(namef, 768, nam_fc_w, 768, nam_fc_b, out_name, OUTD, NODE, OUTD, 768, stream);
  launch_gemm<true>(charf, 100, chr_fc_w, 100, chr_fc_b, out_char, OUTD, NODE, OUTD, 100, stream);

  // ---- nr_gat for relations ----
  run_nr_gat(rel_feat, relw, adj_row, adj_col, ri1, rval,
             r_attn, r_proxy, r_gate, r_bias, nullptr, out_rel,
             segm, segs, attE, newf, outb, lnout, lnproxy, patt, pf, pfg, stream);

  // ---- nr_gat for entities (with img injection) ----
  run_nr_gat(ent_feat, relw, adj_row, adj_col, ri1, rval,
             e_attn, e_proxy, e_gate, e_bias, out_img, out_gph,
             segm, segs, attE, newf, outb, lnout, lnproxy, patt, pf, pfg, stream);
}
